// RNN_18485539242577
// MI455X (gfx1250) — matrix operand order repaired and checked by reading, not compiled, not run on hardware
//
#include <hip/hip_runtime.h>

typedef __attribute__((ext_vector_type(16))) _Float16 v16h;
typedef __attribute__((ext_vector_type(8)))  _Float16 v8h;
typedef __attribute__((ext_vector_type(4)))  _Float16 v4h;
typedef __attribute__((ext_vector_type(8)))  float    v8f;
typedef __attribute__((ext_vector_type(4)))  float    v4f;

#define B_   512
#define T_   2048
#define D_   8
#define H_   64
#define G_   256     // 4*H
#define BT   16      // batch rows per workgroup
#define SA0  104     // padded row stride (halves) for layer0 A staging (K=96 + pad, 16B aligned, conflict-free)
#define SA1  136     // padded row stride (halves) for layer1 A staging (K=128 + pad)
#define SG   260     // gate buffer row stride (floats)

__device__ __forceinline__ float fast_sigmoid(float x) {
    return __builtin_amdgcn_rcpf(1.0f + __expf(-x));
}
__device__ __forceinline__ float fast_tanh(float x) {
    float e = __expf(2.0f * x);
    return (e - 1.0f) * __builtin_amdgcn_rcpf(e + 1.0f);
}

// Build a B-matrix fragment (32x16 f16, lane = column, 16 contiguous K per lane)
// directly from 16 consecutive f32 weights in global memory.
__device__ __forceinline__ v16h frag16_f32(const float* __restrict__ p) {
    v4f a = *(const v4f*)(p);
    v4f b = *(const v4f*)(p + 4);
    v4f c = *(const v4f*)(p + 8);
    v4f d = *(const v4f*)(p + 12);
    v16h r;
    r[0]=(_Float16)a.x;  r[1]=(_Float16)a.y;  r[2]=(_Float16)a.z;  r[3]=(_Float16)a.w;
    r[4]=(_Float16)b.x;  r[5]=(_Float16)b.y;  r[6]=(_Float16)b.z;  r[7]=(_Float16)b.w;
    r[8]=(_Float16)c.x;  r[9]=(_Float16)c.y;  r[10]=(_Float16)c.z; r[11]=(_Float16)c.w;
    r[12]=(_Float16)d.x; r[13]=(_Float16)d.y; r[14]=(_Float16)d.z; r[15]=(_Float16)d.w;
    return r;
}

// Fragment for the K=64..95 block of layer0: first 8 K are w_ih0 row (D=8), rest zero.
__device__ __forceinline__ v16h fragx_f32(const float* __restrict__ p, int khalf) {
    v16h r;
    #pragma unroll
    for (int i = 0; i < 16; ++i) r[i] = (_Float16)0.0f;
    if (khalf == 0) {
        v4f a = *(const v4f*)(p);
        v4f b = *(const v4f*)(p + 4);
        r[0]=(_Float16)a.x; r[1]=(_Float16)a.y; r[2]=(_Float16)a.z; r[3]=(_Float16)a.w;
        r[4]=(_Float16)b.x; r[5]=(_Float16)b.y; r[6]=(_Float16)b.z; r[7]=(_Float16)b.w;
    }
    return r;
}

// Load an A-matrix fragment (16x32 f16) from LDS, contiguous K per lane half:
// lane L: row = L%16; VGPR0-3 hold K = kbase + (L/16)*16 + 0..7,
//                     VGPR4-7 hold K = kbase + (L/16)*16 + 8..15.
__device__ __forceinline__ v16h load_fragA(const _Float16* __restrict__ base, int stride,
                                           int m, int khalf, int kbase) {
    const v8h* p1 = (const v8h*)(base + m * stride + kbase + khalf * 16);
    const v8h* p2 = (const v8h*)(base + m * stride + kbase + khalf * 16 + 8);
    v8h lo = *p1;
    v8h hi = *p2;
    return __builtin_shufflevector(lo, hi, 0,1,2,3,4,5,6,7,8,9,10,11,12,13,14,15);
}

#define WMMA(acc, a, b) \
    acc = __builtin_amdgcn_wmma_f32_16x16x32_f16(false, (a), false, (b), (short)0, (acc), false, false)

__global__ __launch_bounds__(256)
void lstm2_fused_kernel(const float* __restrict__ x,
                        const float* __restrict__ w_ih0, const float* __restrict__ w_hh0,
                        const float* __restrict__ b_ih0, const float* __restrict__ b_hh0,
                        const float* __restrict__ w_ih1, const float* __restrict__ w_hh1,
                        const float* __restrict__ b_ih1, const float* __restrict__ b_hh1,
                        const float* __restrict__ w_out, const float* __restrict__ b_out,
                        float* __restrict__ out)
{
    // LDS: A-staging (f16, padded strides -> conflict-free row reads) + gate buffer (f32)
    __shared__ _Float16 sA0[BT][SA0];   // [ h0(64) | x(8) | pad ]
    __shared__ _Float16 sA1[BT][SA1];   // [ h1(64) | h0_new(64) | pad ]
    __shared__ float    sG[BT][SG];     // gate pre-activations, 256 cols (i|f|g|o)

    const int tid   = threadIdx.x;
    const int lane  = tid & 31;
    const int wv    = tid >> 5;          // wave 0..7
    const int b0    = blockIdx.x * BT;   // batch base for this workgroup
    const int mcol  = lane & 15;         // column within tile / row within A
    const int khalf = lane >> 4;         // K-half selector for fragments

    // ---- resident weight fragments (stay in VGPRs for the whole sequence) ----
    const int nA = wv * 16 + mcol;          // gate column for N-tile A (tiles 0..7)
    const int nB = (wv + 8) * 16 + mcol;    // gate column for N-tile B (tiles 8..15)

    v16h w0a0 = frag16_f32(w_hh0 + nA * H_ +      khalf * 16);
    v16h w0a1 = frag16_f32(w_hh0 + nA * H_ + 32 + khalf * 16);
    v16h w0a2 = fragx_f32 (w_ih0 + nA * D_, khalf);
    v16h w0b0 = frag16_f32(w_hh0 + nB * H_ +      khalf * 16);
    v16h w0b1 = frag16_f32(w_hh0 + nB * H_ + 32 + khalf * 16);
    v16h w0b2 = fragx_f32 (w_ih0 + nB * D_, khalf);

    v16h w1a0 = frag16_f32(w_hh1 + nA * H_ +      khalf * 16);
    v16h w1a1 = frag16_f32(w_hh1 + nA * H_ + 32 + khalf * 16);
    v16h w1a2 = frag16_f32(w_ih1 + nA * H_ +      khalf * 16);
    v16h w1a3 = frag16_f32(w_ih1 + nA * H_ + 32 + khalf * 16);
    v16h w1b0 = frag16_f32(w_hh1 + nB * H_ +      khalf * 16);
    v16h w1b1 = frag16_f32(w_hh1 + nB * H_ + 32 + khalf * 16);
    v16h w1b2 = frag16_f32(w_ih1 + nB * H_ +      khalf * 16);
    v16h w1b3 = frag16_f32(w_ih1 + nB * H_ + 32 + khalf * 16);

    // biases pre-folded into accumulator init (per-lane = per-gate-column)
    const float biasA0 = b_ih0[nA] + b_hh0[nA];
    const float biasB0 = b_ih0[nB] + b_hh0[nB];
    const float biasA1 = b_ih1[nA] + b_hh1[nA];
    const float biasB1 = b_ih1[nB] + b_hh1[nB];

    // ---- init staging: zero h parts + pads, stage x_0 ----
    for (int idx = tid; idx < BT * SA0; idx += 256) ((_Float16*)sA0)[idx] = (_Float16)0.0f;
    for (int idx = tid; idx < BT * SA1; idx += 256) ((_Float16*)sA1)[idx] = (_Float16)0.0f;
    __syncthreads();
    if (tid < 32) {
        int r = tid >> 1, half = tid & 1;
        v4f xv = *(const v4f*)(x + ((size_t)(b0 + r) * T_) * D_ + half * 4);
        v4h xh;
        xh[0]=(_Float16)xv.x; xh[1]=(_Float16)xv.y; xh[2]=(_Float16)xv.z; xh[3]=(_Float16)xv.w;
        *(v4h*)&sA0[r][H_ + half * 4] = xh;
    }

    // per-thread cell state: thread owns rows/units (erow, eu..eu+3) for both layers
    const int erow = tid >> 4;
    const int eu   = (tid & 15) * 4;
    float c0r[4] = {0.f, 0.f, 0.f, 0.f};
    float c1r[4] = {0.f, 0.f, 0.f, 0.f};
    __syncthreads();

    const int grow = khalf * 8;   // C/D layout: VGPR r -> row r (+8 for upper half-wave)

    #pragma unroll 1
    for (int t = 0; t < T_; ++t) {
        // ================= layer 0: gates = [h0|x]*W0^T + b0 =================
        v16h a0 = load_fragA(&sA0[0][0], SA0, mcol, khalf, 0);
        v16h a1 = load_fragA(&sA0[0][0], SA0, mcol, khalf, 32);
        v16h a2 = load_fragA(&sA0[0][0], SA0, mcol, khalf, 64);

        // overlap: fetch next-step x into registers; prefetch t+8 into cache
        const bool havex = (t + 1 < T_);
        v4f xv = {0.f, 0.f, 0.f, 0.f};
        if (tid < 32 && havex) {
            int r = tid >> 1, half = tid & 1;
            xv = *(const v4f*)(x + ((size_t)(b0 + r) * T_ + (t + 1)) * D_ + half * 4);
        }
        if (tid < BT && t + 8 < T_)
            __builtin_prefetch(x + ((size_t)(b0 + tid) * T_ + (t + 8)) * D_, 0, 0);

        v8f accA = {biasA0, biasA0, biasA0, biasA0, biasA0, biasA0, biasA0, biasA0};
        v8f accB = {biasB0, biasB0, biasB0, biasB0, biasB0, biasB0, biasB0, biasB0};
        WMMA(accA, a0, w0a0); WMMA(accA, a1, w0a1); WMMA(accA, a2, w0a2);
        WMMA(accB, a0, w0b0); WMMA(accB, a1, w0b1); WMMA(accB, a2, w0b2);

        #pragma unroll
        for (int r = 0; r < 8; ++r) {
            sG[grow + r][wv * 16 + mcol]         = accA[r];
            sG[grow + r][(wv + 8) * 16 + mcol]   = accB[r];
        }
        __syncthreads();

        // ---- elementwise LSTM update, layer 0 (c in registers) ----
        {
            float hv[4];
            #pragma unroll
            for (int j = 0; j < 4; ++j) {
                int u = eu + j;
                float ig = fast_sigmoid(sG[erow][u]);
                float fg = fast_sigmoid(sG[erow][64 + u]);
                float gg = fast_tanh   (sG[erow][128 + u]);
                float og = fast_sigmoid(sG[erow][192 + u]);
                float c  = fg * c0r[j] + ig * gg;
                c0r[j]   = c;
                hv[j]    = og * fast_tanh(c);
            }
            v4h hh;
            hh[0]=(_Float16)hv[0]; hh[1]=(_Float16)hv[1]; hh[2]=(_Float16)hv[2]; hh[3]=(_Float16)hv[3];
            *(v4h*)&sA0[erow][eu]       = hh;   // next step's recurrent input
            *(v4h*)&sA1[erow][H_ + eu]  = hh;   // this step's layer-1 input
        }
        if (tid < 32 && havex) {
            int r = tid >> 1, half = tid & 1;
            v4h xh;
            xh[0]=(_Float16)xv.x; xh[1]=(_Float16)xv.y; xh[2]=(_Float16)xv.z; xh[3]=(_Float16)xv.w;
            *(v4h*)&sA0[r][H_ + half * 4] = xh;
        }
        __syncthreads();

        // ================= layer 1: gates = [h1|h0_new]*W1^T + b1 =================
        v16h q0 = load_fragA(&sA1[0][0], SA1, mcol, khalf, 0);
        v16h q1 = load_fragA(&sA1[0][0], SA1, mcol, khalf, 32);
        v16h q2 = load_fragA(&sA1[0][0], SA1, mcol, khalf, 64);
        v16h q3 = load_fragA(&sA1[0][0], SA1, mcol, khalf, 96);

        v8f acc1A = {biasA1, biasA1, biasA1, biasA1, biasA1, biasA1, biasA1, biasA1};
        v8f acc1B = {biasB1, biasB1, biasB1, biasB1, biasB1, biasB1, biasB1, biasB1};
        WMMA(acc1A, q0, w1a0); WMMA(acc1A, q1, w1a1); WMMA(acc1A, q2, w1a2); WMMA(acc1A, q3, w1a3);
        WMMA(acc1B, q0, w1b0); WMMA(acc1B, q1, w1b1); WMMA(acc1B, q2, w1b2); WMMA(acc1B, q3, w1b3);

        #pragma unroll
        for (int r = 0; r < 8; ++r) {
            sG[grow + r][wv * 16 + mcol]       = acc1A[r];
            sG[grow + r][(wv + 8) * 16 + mcol] = acc1B[r];
        }
        __syncthreads();

        // ---- elementwise LSTM update, layer 1 ----
        {
            float hv[4];
            #pragma unroll
            for (int j = 0; j < 4; ++j) {
                int u = eu + j;
                float ig = fast_sigmoid(sG[erow][u]);
                float fg = fast_sigmoid(sG[erow][64 + u]);
                float gg = fast_tanh   (sG[erow][128 + u]);
                float og = fast_sigmoid(sG[erow][192 + u]);
                float c  = fg * c1r[j] + ig * gg;
                c1r[j]   = c;
                hv[j]    = og * fast_tanh(c);
            }
            v4h hh;
            hh[0]=(_Float16)hv[0]; hh[1]=(_Float16)hv[1]; hh[2]=(_Float16)hv[2]; hh[3]=(_Float16)hv[3];
            *(v4h*)&sA1[erow][eu] = hh;   // next step's layer-1 recurrent input
        }
        __syncthreads();
    }

    // ================= output: h2[:, -1, :] @ w_out^T + b_out =================
    if (tid < BT) {
        float s = 0.0f;
        #pragma unroll 8
        for (int j = 0; j < H_; ++j)
            s += (float)sA1[tid][j] * w_out[j];
        out[b0 + tid] = s + b_out[0];
    }
}

extern "C" void kernel_launch(void* const* d_in, const int* in_sizes, int n_in,
                              void* d_out, int out_size, void* d_ws, size_t ws_size,
                              hipStream_t stream) {
    const float* x     = (const float*)d_in[0];
    const float* w_ih0 = (const float*)d_in[1];
    const float* w_hh0 = (const float*)d_in[2];
    const float* b_ih0 = (const float*)d_in[3];
    const float* b_hh0 = (const float*)d_in[4];
    const float* w_ih1 = (const float*)d_in[5];
    const float* w_hh1 = (const float*)d_in[6];
    const float* b_ih1 = (const float*)d_in[7];
    const float* b_hh1 = (const float*)d_in[8];
    const float* w_out = (const float*)d_in[9];
    const float* b_out = (const float*)d_in[10];
    float* out = (float*)d_out;

    lstm2_fused_kernel<<<dim3(B_ / BT), dim3(256), 0, stream>>>(
        x, w_ih0, w_hh0, b_ih0, b_hh0, w_ih1, w_hh1, b_ih1, b_hh1, w_out, b_out, out);
}
